// Generator_40965398069405
// MI455X (gfx1250) — compile-verified, matrix operand order checked by reading, not run on hardware
//
#include <hip/hip_runtime.h>

#define NUM_ITEMS 50000
#define EMB_DIM   256
#define BATCH     4096
#define TOPK      10

#define M_TILE          128                       // batch rows per block
#define ROW_BLOCKS      (BATCH / M_TILE)          // 32
#define NSPLIT          5
#define TILES_PER_SPLIT (NUM_ITEMS / 16 / NSPLIT) // 625 tiles of 16 items
#define CAP             48                        // candidate buffer capacity per row
#define TRIG            32                        // compaction trigger (CAP-TRIG >= 16/tile)
#define NCAND           (NSPLIT * 16)             // 80 candidates per row total
#define SENT            0x7FFFFFFF

typedef __attribute__((ext_vector_type(16))) __bf16 v16bf;
typedef __attribute__((ext_vector_type(8)))  __bf16 v8bf;
typedef __attribute__((ext_vector_type(8)))  float  v8f;
typedef __attribute__((ext_vector_type(4)))  unsigned v4u;
typedef __attribute__((ext_vector_type(8)))  int      v8i;
typedef __attribute__((ext_vector_type(4)))  int      v4i;

// ---------------------------------------------------------------------------
// TDM: issue a tensor_load_to_lds of one contiguous 16-row x 512-byte bf16
// tile (described as a 2D tile: 16 rows x 64 units of 8 bytes). D# bitfields
// per CDNA5 ISA ch.8. Groups 2/3 are unused (2D tensor) and passed as zero.
// This toolchain exposes the 6-arg builtin (g0, g1, g2, g3, extra, cpol).
// ---------------------------------------------------------------------------
__device__ __forceinline__ void tdm_load_tile(const void* gsrc, unsigned lds_off) {
#if defined(__HIP_DEVICE_COMPILE__)
  const unsigned long long ga = (unsigned long long)(uintptr_t)gsrc;
  v4u g0;
  g0.x = 1u;                                             // count=1, user D#
  g0.y = lds_off;                                        // lds_addr (bytes)
  g0.z = (unsigned)ga;                                   // global_addr[31:0]
  g0.w = (unsigned)((ga >> 32) & 0x01FFFFFFull)          // global_addr[56:32]
       | (2u << 30);                                     // type = 2 ("image")
  v8i g1;
  g1[0] = (int)(3u << 16);   // wg_mask=0, data_size=3 (8B), no flags, no pad
  g1[1] = (int)(64u << 16);  // [63:48] tensor_dim0[15:0] = 64
  g1[2] = (int)(16u << 16);  // [79:64] tensor_dim0[31:16]=0, [95:80] tensor_dim1[15:0]=16
  g1[3] = (int)(64u << 16);  // [111:96] tensor_dim1[31:16]=0, [127:112] tile_dim0=64
  g1[4] = (int)16u;          // [143:128] tile_dim1=16, [159:144] tile_dim2=0
  g1[5] = (int)64;           // [191:160] tensor_dim0_stride[31:0] = 64
  g1[6] = 0;                 // stride0[47:32]=0, stride1[15:0]=0
  g1[7] = 0;                 // stride1[47:16]=0
  const v4i z4 = {0, 0, 0, 0};
  const v8i z8 = {0, 0, 0, 0, 0, 0, 0, 0};
  __builtin_amdgcn_tensor_load_to_lds(g0, g1, z4, z4, z8, 0);
#else
  (void)gsrc; (void)lds_off;
#endif
}

__device__ __forceinline__ void tdm_wait(int n) {
#if defined(__HIP_DEVICE_COMPILE__)
  if (n) __builtin_amdgcn_s_wait_tensorcnt(1);
  else   __builtin_amdgcn_s_wait_tensorcnt(0);
#else
  (void)n;
#endif
}

// Load a 16x32 bf16 A/B fragment from a row-major [16][256] bf16 LDS tile.
// ISA layout (16-bit A 16x32): lanes 0-15 hold K {k0..k0+7, k0+16..k0+23},
// lanes 16-31 hold K {k0+8..k0+15, k0+24..k0+31}. B (32x16) mirrors this with
// N = lane&15, so the same loader works when 'rowbase' is the item row.
__device__ __forceinline__ v16bf frag_ld(const unsigned short* rowbase, int lane, int k0) {
  const int half8 = (lane & 16) >> 1;             // 0 or 8 elements
  const v8bf* p = (const v8bf*)(rowbase + k0 + half8);
  v8bf lo = p[0];                                 // K = k0+half8 .. +8
  v8bf hi = p[2];                                 // K = k0+16+half8 .. +8
  return __builtin_shufflevector(lo, hi, 0,1,2,3,4,5,6,7,8,9,10,11,12,13,14,15);
}

// Exact top-16 compaction of up to CAP (score,idx) entries via wave shuffles.
// Sets cnt=16 and thr = 16th-best score.
__device__ __forceinline__ void compact_row(float* bS, int* bI, unsigned* cntp,
                                            float* thrp, int lane) {
  int n = (int)*cntp; n = n > CAP ? CAP : n;
  float s0 = -__builtin_inff(), s1 = -__builtin_inff();
  int i0 = SENT, i1 = SENT;
  if (lane < n)      { s0 = bS[lane];      i0 = bI[lane]; }
  if (lane + 32 < n) { s1 = bS[lane + 32]; i1 = bI[lane + 32]; }
  float keepS = -__builtin_inff(); int keepI = SENT;
#pragma unroll 1
  for (int sel = 0; sel < 16; ++sel) {
    float bs; int bi, bw;
    if (s1 > s0 || (s1 == s0 && i1 < i0)) { bs = s1; bi = i1; bw = 1; }
    else                                  { bs = s0; bi = i0; bw = 0; }
    int bl = lane;
#pragma unroll
    for (int off = 16; off >= 1; off >>= 1) {
      float os = __shfl_xor(bs, off, 32);
      int   oi = __shfl_xor(bi, off, 32);
      int   ol = __shfl_xor(bl, off, 32);
      int   ow = __shfl_xor(bw, off, 32);
      if (os > bs || (os == bs && oi < bi)) { bs = os; bi = oi; bl = ol; bw = ow; }
    }
    if (lane == bl) {
      if (bw) { s1 = -__builtin_inff(); i1 = SENT; }
      else    { s0 = -__builtin_inff(); i0 = SENT; }
    }
    if (lane == sel) { keepS = bs; keepI = bi; }
    if (sel == 15 && lane == 0) *thrp = bs;
  }
  if (lane < 16) { bS[lane] = keepS; bI[lane] = keepI; }
  if (lane == 0) *cntp = 16u;
}

// f32 -> bf16 conversion of the item matrix (one-time, bandwidth bound).
__global__ void cvt_items_kernel(const float* __restrict__ src,
                                 unsigned short* __restrict__ dst) {
  const size_t base = ((size_t)blockIdx.x * 256 + threadIdx.x) * 16;
  const float4* s = (const float4*)(src + base);
  unsigned o[8];
#pragma unroll
  for (int q = 0; q < 4; ++q) {
    const float4 v = s[q];
    o[2 * q + 0] = (__float_as_uint(v.x) >> 16) | (__float_as_uint(v.y) & 0xFFFF0000u);
    o[2 * q + 1] = (__float_as_uint(v.z) >> 16) | (__float_as_uint(v.w) & 0xFFFF0000u);
  }
  uint4* d = (uint4*)(dst + base);
  d[0] = make_uint4(o[0], o[1], o[2], o[3]);
  d[1] = make_uint4(o[4], o[5], o[6], o[7]);
}

// ---------------------------------------------------------------------------
// TDM-staged scoring kernel: wave 0 double-buffers the bf16 B tile via the
// Tensor Data Mover while all 8 waves run WMMA + streaming top-select.
// ---------------------------------------------------------------------------
__global__ void score_topk_tdm_kernel(const int* __restrict__ users,
                                      const float* __restrict__ emb_users,
                                      const unsigned short* __restrict__ items_bf,
                                      const float* __restrict__ bias_items,
                                      const unsigned char* __restrict__ bought,
                                      int* __restrict__ candI_g) {
  extern __shared__ __align__(16) char smem[];
  unsigned short* Abuf = (unsigned short*)smem;            // 128*256 bf16 = 64 KB
  unsigned short* Bbuf = Abuf + M_TILE * EMB_DIM;          // 2*16*256 bf16 = 16 KB
  float*    candS = (float*)(Bbuf + 2 * 16 * EMB_DIM);     // 128*48 f32   = 24 KB
  int*      candI = (int*)(candS + M_TILE * CAP);          //              = 24 KB
  unsigned* cnt   = (unsigned*)(candI + M_TILE * CAP);     // 128
  float*    thr   = (float*)(cnt + M_TILE);                // 128
  int*      uid   = (int*)(thr + M_TILE);                  // 128

  const int tid  = threadIdx.x;
  const int w    = tid >> 5;
  const int lane = tid & 31;
  const int rb      = blockIdx.x / NSPLIT;
  const int isplit  = blockIdx.x % NSPLIT;
  const int rowBase = rb * M_TILE;
  const unsigned bbase = (unsigned)(uintptr_t)Bbuf;        // LDS byte offset
  const int tile0 = isplit * TILES_PER_SPLIT;

  if (tid < M_TILE) {
    uid[tid] = users[rowBase + tid];
    cnt[tid] = 0u;
    thr[tid] = -__builtin_inff();
  }
  // Kick off the DMA for tile 0 while the A panel is being gathered.
  if (w == 0)
    tdm_load_tile(items_bf + (size_t)tile0 * 16 * EMB_DIM, bbase);
  __syncthreads();
  // Gather + convert the A panel (128 users x 256) to bf16 in LDS.
  for (int i = tid; i < M_TILE * EMB_DIM; i += 256) {
    const int r = i >> 8, k = i & 255;
    const float v = emb_users[(size_t)uid[r] * EMB_DIM + k];
    Abuf[i] = (unsigned short)(__float_as_uint(v) >> 16);
  }

  const unsigned short* Arow = Abuf + (size_t)(w * 16 + (lane & 15)) * EMB_DIM;
  const unsigned short* Brow0 = Bbuf + (size_t)(lane & 15) * EMB_DIM;
  const int halfAdd = (lane & 16) ? 8 : 0;   // C layout: lanes 16-31 hold M = j+8

  const unsigned char* mrow[8];
#pragma unroll
  for (int j = 0; j < 8; ++j)
    mrow[j] = bought + (size_t)uid[w * 16 + j + halfAdd] * NUM_ITEMS;

  for (int t = 0; t < TILES_PER_SPLIT; ++t) {
    if (w == 0) {
      if (t + 1 < TILES_PER_SPLIT) {
        // Issue DMA for tile t+1 into the other buffer, then wait for tile t.
        tdm_load_tile(items_bf + (size_t)(tile0 + t + 1) * 16 * EMB_DIM,
                      bbase + (unsigned)(((t + 1) & 1) * 16 * EMB_DIM * 2));
        tdm_wait(1);
      } else {
        tdm_wait(0);
      }
    }
    __syncthreads();   // buffer (t&1) is ready for all waves

    const int ibase = (tile0 + t) * 16;
    const int col = ibase + (lane & 15);
    const unsigned short* Brow = Brow0 + (t & 1) * 16 * EMB_DIM;

    // Issue slow-path operands up front; latency hides under the 8 WMMAs.
    unsigned char mb[8];
#pragma unroll
    for (int j = 0; j < 8; ++j) mb[j] = mrow[j][col];
    const float bsv = bias_items[col];
    float th[8];
#pragma unroll
    for (int j = 0; j < 8; ++j) th[j] = thr[w * 16 + j + halfAdd];

    // Software-pipelined K loop (fragments for k+1 in flight during WMMA k).
    v8f acc = {0.f, 0.f, 0.f, 0.f, 0.f, 0.f, 0.f, 0.f};
    v16bf av = frag_ld(Arow, lane, 0);
    v16bf bv = frag_ld(Brow, lane, 0);
#pragma unroll
    for (int k0 = 0; k0 < EMB_DIM - 32; k0 += 32) {
      v16bf an = frag_ld(Arow, lane, k0 + 32);
      v16bf bn = frag_ld(Brow, lane, k0 + 32);
      acc = __builtin_amdgcn_wmma_f32_16x16x32_bf16(false, av, false, bv,
                                                    (short)0, acc, false, false);
      av = an; bv = bn;
    }
    acc = __builtin_amdgcn_wmma_f32_16x16x32_bf16(false, av, false, bv,
                                                  (short)0, acc, false, false);

#pragma unroll
    for (int j = 0; j < 8; ++j) {
      const int widx = w * 16 + j + halfAdd;
      const float sc = acc[j] + bsv;
      if (!mb[j] && sc > th[j]) {
        const unsigned pos = atomicAdd(&cnt[widx], 1u);
        if (pos < CAP) { candS[widx * CAP + pos] = sc; candI[widx * CAP + pos] = col; }
      }
    }

    const unsigned nb = (lane < 16) ? cnt[w * 16 + lane] : 0u;
    if (__builtin_amdgcn_ballot_w32(nb >= TRIG)) {
      for (int r = 0; r < 16; ++r) {
        const int widx = w * 16 + r;
        if (cnt[widx] >= TRIG)
          compact_row(candS + widx * CAP, candI + widx * CAP, cnt + widx, thr + widx, lane);
      }
    }
    __syncthreads();   // all waves done with buffer (t&1) before it is refilled
  }

  for (int r = 0; r < 16; ++r) {
    const int widx = w * 16 + r;
    if (cnt[widx] > 16u)
      compact_row(candS + widx * CAP, candI + widx * CAP, cnt + widx, thr + widx, lane);
  }
  for (int r = 0; r < 16; ++r) {
    const int widx = w * 16 + r;
    unsigned n = cnt[widx]; if (n > 16u) n = 16u;
    if (lane < 16) {
      int idx = SENT;
      if ((unsigned)lane < n) idx = candI[widx * CAP + lane];
      candI_g[(size_t)(rowBase + widx) * NCAND + isplit * 16 + lane] = idx;
    }
  }
}

// ---------------------------------------------------------------------------
// Fallback scoring kernel (cooperative f32->bf16 staging), used if d_ws is too
// small to hold the bf16 item matrix. Same algorithm as round 2.
// ---------------------------------------------------------------------------
__global__ void score_topk_fallback_kernel(const int* __restrict__ users,
                                           const float* __restrict__ emb_users,
                                           const float* __restrict__ emb_items,
                                           const float* __restrict__ bias_items,
                                           const unsigned char* __restrict__ bought,
                                           int* __restrict__ candI_g) {
  extern __shared__ __align__(16) char smem[];
  unsigned short* Abuf = (unsigned short*)smem;
  unsigned short* Bbuf = Abuf + M_TILE * EMB_DIM;
  float*    candS = (float*)(Bbuf + 16 * EMB_DIM);
  int*      candI = (int*)(candS + M_TILE * CAP);
  unsigned* cnt   = (unsigned*)(candI + M_TILE * CAP);
  float*    thr   = (float*)(cnt + M_TILE);
  int*      uid   = (int*)(thr + M_TILE);

  const int tid  = threadIdx.x;
  const int w    = tid >> 5;
  const int lane = tid & 31;
  const int rb      = blockIdx.x / NSPLIT;
  const int isplit  = blockIdx.x % NSPLIT;
  const int rowBase = rb * M_TILE;

  if (tid < M_TILE) {
    uid[tid] = users[rowBase + tid];
    cnt[tid] = 0u;
    thr[tid] = -__builtin_inff();
  }
  __syncthreads();
  for (int i = tid; i < M_TILE * EMB_DIM; i += 256) {
    const int r = i >> 8, k = i & 255;
    const float v = emb_users[(size_t)uid[r] * EMB_DIM + k];
    Abuf[i] = (unsigned short)(__float_as_uint(v) >> 16);
  }

  const unsigned short* Arow = Abuf + (size_t)(w * 16 + (lane & 15)) * EMB_DIM;
  const unsigned short* Brow = Bbuf + (size_t)(lane & 15) * EMB_DIM;
  const int halfAdd = (lane & 16) ? 8 : 0;

  const unsigned char* mrow[8];
#pragma unroll
  for (int j = 0; j < 8; ++j)
    mrow[j] = bought + (size_t)uid[w * 16 + j + halfAdd] * NUM_ITEMS;

  for (int t = 0; t < TILES_PER_SPLIT; ++t) {
    const int ibase = (isplit * TILES_PER_SPLIT + t) * 16;
    __syncthreads();
    {
      const int br = tid >> 4, c0 = (tid & 15) << 4;
      const float* src = emb_items + (size_t)(ibase + br) * EMB_DIM + c0;
      unsigned* dst = (unsigned*)(Bbuf + br * EMB_DIM + c0);
#pragma unroll
      for (int q = 0; q < 4; ++q) {
        const float4 v = ((const float4*)src)[q];
        dst[2 * q + 0] = (__float_as_uint(v.x) >> 16) | (__float_as_uint(v.y) & 0xFFFF0000u);
        dst[2 * q + 1] = (__float_as_uint(v.z) >> 16) | (__float_as_uint(v.w) & 0xFFFF0000u);
      }
      if (t + 1 < TILES_PER_SPLIT) __builtin_prefetch(src + 16 * EMB_DIM, 0, 1);
    }
    __syncthreads();

    const int col = ibase + (lane & 15);
    unsigned char mb[8];
#pragma unroll
    for (int j = 0; j < 8; ++j) mb[j] = mrow[j][col];
    const float bsv = bias_items[col];
    float th[8];
#pragma unroll
    for (int j = 0; j < 8; ++j) th[j] = thr[w * 16 + j + halfAdd];

    v8f acc = {0.f, 0.f, 0.f, 0.f, 0.f, 0.f, 0.f, 0.f};
    v16bf av = frag_ld(Arow, lane, 0);
    v16bf bv = frag_ld(Brow, lane, 0);
#pragma unroll
    for (int k0 = 0; k0 < EMB_DIM - 32; k0 += 32) {
      v16bf an = frag_ld(Arow, lane, k0 + 32);
      v16bf bn = frag_ld(Brow, lane, k0 + 32);
      acc = __builtin_amdgcn_wmma_f32_16x16x32_bf16(false, av, false, bv,
                                                    (short)0, acc, false, false);
      av = an; bv = bn;
    }
    acc = __builtin_amdgcn_wmma_f32_16x16x32_bf16(false, av, false, bv,
                                                  (short)0, acc, false, false);

#pragma unroll
    for (int j = 0; j < 8; ++j) {
      const int widx = w * 16 + j + halfAdd;
      const float sc = acc[j] + bsv;
      if (!mb[j] && sc > th[j]) {
        const unsigned pos = atomicAdd(&cnt[widx], 1u);
        if (pos < CAP) { candS[widx * CAP + pos] = sc; candI[widx * CAP + pos] = col; }
      }
    }

    const unsigned nb = (lane < 16) ? cnt[w * 16 + lane] : 0u;
    if (__builtin_amdgcn_ballot_w32(nb >= TRIG)) {
      for (int r = 0; r < 16; ++r) {
        const int widx = w * 16 + r;
        if (cnt[widx] >= TRIG)
          compact_row(candS + widx * CAP, candI + widx * CAP, cnt + widx, thr + widx, lane);
      }
    }
  }

  for (int r = 0; r < 16; ++r) {
    const int widx = w * 16 + r;
    if (cnt[widx] > 16u)
      compact_row(candS + widx * CAP, candI + widx * CAP, cnt + widx, thr + widx, lane);
  }
  for (int r = 0; r < 16; ++r) {
    const int widx = w * 16 + r;
    unsigned n = cnt[widx]; if (n > 16u) n = 16u;
    if (lane < 16) {
      int idx = SENT;
      if ((unsigned)lane < n) idx = candI[widx * CAP + lane];
      candI_g[(size_t)(rowBase + widx) * NCAND + isplit * 16 + lane] = idx;
    }
  }
}

// FP32-exact rescoring of the <=80 candidates per row + top-10 selection with
// jax.lax.top_k tie-breaking (lower index wins on equal score).
__global__ void rescore_topk_kernel(const int* __restrict__ users,
                                    const float* __restrict__ emb_users,
                                    const float* __restrict__ emb_items,
                                    const float* __restrict__ bias_items,
                                    const int* __restrict__ candI_g,
                                    int* __restrict__ out) {
  const int w = threadIdx.x >> 5, lane = threadIdx.x & 31;
  const int gRow = blockIdx.x * 8 + w;
  const int u = users[gRow];
  float uv[8];
#pragma unroll
  for (int q = 0; q < 8; ++q) uv[q] = emb_users[(size_t)u * EMB_DIM + lane + 32 * q];

  float rs[3] = {-__builtin_inff(), -__builtin_inff(), -__builtin_inff()};
  int   ri[3] = {SENT, SENT, SENT};
#pragma unroll 1
  for (int c = 0; c < NCAND; ++c) {
    const int idx = candI_g[(size_t)gRow * NCAND + c];
    float sc = -__builtin_inff();
    if (idx != SENT) {
      const float* ip = emb_items + (size_t)idx * EMB_DIM;
      float a = 0.f;
#pragma unroll
      for (int q = 0; q < 8; ++q) a += uv[q] * ip[lane + 32 * q];
#pragma unroll
      for (int off = 16; off >= 1; off >>= 1) a += __shfl_xor(a, off, 32);
      sc = a + bias_items[idx];
    }
    if ((c & 31) == lane) { rs[c >> 5] = sc; ri[c >> 5] = idx; }
  }
#pragma unroll 1
  for (int sel = 0; sel < TOPK; ++sel) {
    float bs = rs[0]; int bi = ri[0]; int bz = 0;
    if (rs[1] > bs || (rs[1] == bs && ri[1] < bi)) { bs = rs[1]; bi = ri[1]; bz = 1; }
    if (rs[2] > bs || (rs[2] == bs && ri[2] < bi)) { bs = rs[2]; bi = ri[2]; bz = 2; }
    int bl = lane;
#pragma unroll
    for (int off = 16; off >= 1; off >>= 1) {
      float os = __shfl_xor(bs, off, 32);
      int   oi = __shfl_xor(bi, off, 32);
      int   ol = __shfl_xor(bl, off, 32);
      int   oz = __shfl_xor(bz, off, 32);
      if (os > bs || (os == bs && oi < bi)) { bs = os; bi = oi; bl = ol; bz = oz; }
    }
    if (lane == bl) { rs[bz] = -__builtin_inff(); ri[bz] = SENT; }
    if (lane == 0) out[gRow * TOPK + sel] = bi;
  }
}

extern "C" void kernel_launch(void* const* d_in, const int* in_sizes, int n_in,
                              void* d_out, int out_size, void* d_ws, size_t ws_size,
                              hipStream_t stream) {
  const int*           users     = (const int*)d_in[0];
  // d_in[1] = k (always TOPK)
  const float*         emb_users = (const float*)d_in[2];
  const float*         emb_items = (const float*)d_in[3];
  const float*         bias      = (const float*)d_in[4];
  const unsigned char* bought    = (const unsigned char*)d_in[5];
  int* out = (int*)d_out;

  const size_t itemsBfBytes = (size_t)NUM_ITEMS * EMB_DIM * 2;  // 25.6 MB
  const size_t candBytes    = (size_t)BATCH * NCAND * 4;        // 1.31 MB
  const size_t candLds  = (size_t)M_TILE * CAP * 4 * 2 + (size_t)M_TILE * 12;

  if (ws_size >= itemsBfBytes + candBytes) {
    unsigned short* items_bf = (unsigned short*)d_ws;
    int* candI_g = (int*)((char*)d_ws + itemsBfBytes);
    const size_t smem = (size_t)M_TILE * EMB_DIM * 2      // A bf16
                      + (size_t)2 * 16 * EMB_DIM * 2      // B bf16 double buffer
                      + candLds;
    cvt_items_kernel<<<dim3(NUM_ITEMS * EMB_DIM / (256 * 16)), dim3(256), 0, stream>>>(
        emb_items, items_bf);
    score_topk_tdm_kernel<<<dim3(ROW_BLOCKS * NSPLIT), dim3(256), smem, stream>>>(
        users, emb_users, items_bf, bias, bought, candI_g);
    rescore_topk_kernel<<<dim3(BATCH / 8), dim3(256), 0, stream>>>(
        users, emb_users, emb_items, bias, candI_g, out);
  } else {
    int* candI_g = (int*)d_ws;
    const size_t smem = (size_t)M_TILE * EMB_DIM * 2
                      + (size_t)16 * EMB_DIM * 2
                      + candLds;
    score_topk_fallback_kernel<<<dim3(ROW_BLOCKS * NSPLIT), dim3(256), smem, stream>>>(
        users, emb_users, emb_items, bias, bought, candI_g);
    rescore_topk_kernel<<<dim3(BATCH / 8), dim3(256), 0, stream>>>(
        users, emb_users, emb_items, bias, candI_g, out);
  }
}
